// LSTMModel_17995912970658
// MI455X (gfx1250) — matrix operand order repaired and checked by reading, not compiled, not run on hardware
//
#include <hip/hip_runtime.h>

// =====================================================================
// 20-layer LSTM, H=128, B=128, T=1024 on MI455X (gfx1250).
//
// One persistent workgroup per layer (20 WGs on 20 WGPs), wavefront-
// pipelined via produced/consumed flags; inter-layer activations as
// bf16 ring buffers (L2-resident); Wih/Whh/h in LDS (bf16, 136-elem
// row stride: 16B-aligned for ds_load_b128, 68-dword stride -> 16
// lanes on distinct banks); gates via v_wmma_f32_16x16x32_bf16 with
// f32 accumulation (C starts at inline 0, bias folded into epilogue);
// cell state lives in VGPRs.
// =====================================================================

#define T_SEQ   1024
#define BATCH   128
#define HID     128
#define G4H     512          // 4*H
#define NLAYER  20
#define WSTRIDE 136          // LDS row stride (bf16 elems): 272B, 16B-aligned, bank-clean
#define HSTRIDE 136

typedef __bf16 bf16_t;
typedef __attribute__((ext_vector_type(16))) __bf16        v16bf;
typedef __attribute__((ext_vector_type(8)))  float         v8f;
typedef __attribute__((ext_vector_type(4)))  unsigned int  v4u;
typedef __attribute__((ext_vector_type(8)))  unsigned int  v8u;

// ---------------- fast activations (v_exp_f32 + v_rcp_f32) ----------------
__device__ __forceinline__ float fast_sigmoid(float x) {
    x = fminf(fmaxf(x, -30.0f), 30.0f);
    return __builtin_amdgcn_rcpf(1.0f + __expf(-x));
}
__device__ __forceinline__ float fast_tanh(float x) {
    x = fminf(fmaxf(x, -15.0f), 15.0f);
    float e = __expf(2.0f * x);
    return (e - 1.0f) * __builtin_amdgcn_rcpf(e + 1.0f);
}

// ---------------- flag helpers (agent-scope acquire/release) ----------------
__device__ __forceinline__ void wait_ge(int* p, int v) {
    while (__hip_atomic_load(p, __ATOMIC_ACQUIRE, __HIP_MEMORY_SCOPE_AGENT) < v)
        __builtin_amdgcn_s_sleep(2);
}
__device__ __forceinline__ void signal(int* p, int v) {
    __hip_atomic_store(p, v, __ATOMIC_RELEASE, __HIP_MEMORY_SCOPE_AGENT);
}

__device__ __forceinline__ v16bf combine16(v4u lo, v4u hi) {
    v8u v = __builtin_shufflevector(lo, hi, 0, 1, 2, 3, 4, 5, 6, 7);
    return __builtin_bit_cast(v16bf, v);
}

// ---------------- WMMA fragment loaders ----------------
// A (16x32 bf16): lane<16 row M=lane, elems[0..7]=K0..K0+7, [8..15]=K0+16..23;
//                 lane>=16 same row, K shifted +8. Two 16B runs per lane.
__device__ __forceinline__ v16bf load_a_bf16(const bf16_t* __restrict__ base,
                                             int stride, int lane, int K0) {
    const int m  = lane & 15;
    const int s1 = K0 + ((lane & 16) ? 8 : 0);
    const bf16_t* p = base + m * stride + s1;       // 16B aligned
    v4u lo = *(const v4u*)p;
    v4u hi = *(const v4u*)(p + 16);
    return combine16(lo, hi);
}
// Layer-0 path: f32 source, convert to bf16 and force materialization so the
// f32 temporaries die before the WMMA region (prevents register spills).
__device__ __forceinline__ v16bf load_a_f32cvt(const float* __restrict__ base,
                                               int lane, int K0) {
    const int m  = lane & 15;
    const int s1 = K0 + ((lane & 16) ? 8 : 0);
    const float* p = base + m * HID + s1;           // 32B aligned
    float4 f0 = *(const float4*)(p);
    float4 f1 = *(const float4*)(p + 4);
    float4 f2 = *(const float4*)(p + 16);
    float4 f3 = *(const float4*)(p + 20);
    v16bf a;
    a[0]=(bf16_t)f0.x;  a[1]=(bf16_t)f0.y;  a[2]=(bf16_t)f0.z;  a[3]=(bf16_t)f0.w;
    a[4]=(bf16_t)f1.x;  a[5]=(bf16_t)f1.y;  a[6]=(bf16_t)f1.z;  a[7]=(bf16_t)f1.w;
    a[8]=(bf16_t)f2.x;  a[9]=(bf16_t)f2.y;  a[10]=(bf16_t)f2.z; a[11]=(bf16_t)f2.w;
    a[12]=(bf16_t)f3.x; a[13]=(bf16_t)f3.y; a[14]=(bf16_t)f3.z; a[15]=(bf16_t)f3.w;
    v8u t = __builtin_bit_cast(v8u, a);
    asm volatile("" : "+v"(t));                     // register barrier
    return __builtin_bit_cast(v16bf, t);
}
// B (32x16 bf16): lane<16 col N=lane, elems[0..7]=K0..K0+7, [8..15]=K0+16..23;
// lane>=16 K shifted +8. B[k][n] = W[n][k] -> two 16B runs of row n of W.
__device__ __forceinline__ v16bf load_b_lds(const bf16_t* __restrict__ W,
                                            int ncolBase, int lane, int K0) {
    const int n  = ncolBase + (lane & 15);
    const int ks = K0 + ((lane & 16) ? 8 : 0);
    const bf16_t* p = W + n * WSTRIDE + ks;         // 16B aligned
    v4u lo = *(const v4u*)p;
    v4u hi = *(const v4u*)(p + 16);
    return combine16(lo, hi);
}

// ---------------- embedding gather ----------------
__global__ void embed_kernel(const int* __restrict__ x,
                             const float* __restrict__ emb,
                             float* __restrict__ outEmb) {
    const size_t idx = ((size_t)blockIdx.x * blockDim.x + threadIdx.x) * 4;
    const int row = (int)(idx >> 7);     // t*B + b
    const int h   = (int)(idx & 127);
    const int tok = x[row];
    const float4 v = *(const float4*)(emb + (size_t)tok * HID + h);
    *(float4*)(outEmb + idx) = v;
}

__global__ void zero_flags_kernel(int* __restrict__ f) { f[threadIdx.x] = 0; }

// ---------------- persistent pipelined LSTM ----------------
__global__ __launch_bounds__(512, 1)
void lstm_pipeline(const float* __restrict__ embIn,
                   const float* __restrict__ Wih,
                   const float* __restrict__ Whh,
                   const float* __restrict__ bih,
                   const float* __restrict__ bhh,
                   float* __restrict__ outFinal,
                   bf16_t* __restrict__ rings,       // bf16 activations
                   int ringMask,                     // ringSteps-1 (pow2)
                   int* __restrict__ flags) {
    extern __shared__ __align__(16) char smemRaw[];
    bf16_t* sWih = (bf16_t*)smemRaw;             // [512][WSTRIDE]
    bf16_t* sWhh = sWih + G4H * WSTRIDE;         // [512][WSTRIDE]
    bf16_t* sH   = sWhh + G4H * WSTRIDE;         // [128][HSTRIDE] h_{t-1}

    const int l    = blockIdx.x;
    const int tid  = threadIdx.x;
    const int lane = tid & 31;
    const int wave = tid >> 5;                   // 16 waves
    const int r    = wave >> 1;                  // batch tile row 0..7
    const int c0   = (wave & 1) * 4;             // first hidden tile col
    const bool firstLayer = (l == 0);
    const bool lastLayer  = (l == NLAYER - 1);
    const int ringSteps   = ringMask + 1;

    const float* Wih_l = Wih + (size_t)l * G4H * HID;
    const float* Whh_l = Whh + (size_t)l * G4H * HID;
    const float* b1    = bih + l * G4H;
    const float* b2    = bhh + l * G4H;

    int* produced = flags;        // produced[l] = steps completed by layer l
    int* consumed = flags + 32;   // consumed[l] = steps read by layer l

    // --- one-time: stage weights (f32 -> bf16) into LDS; zero h state ---
    for (int i = tid; i < G4H * HID; i += 512) {
        const int row = i >> 7, k = i & 127;
        sWih[row * WSTRIDE + k] = (bf16_t)Wih_l[i];
        sWhh[row * WSTRIDE + k] = (bf16_t)Whh_l[i];
    }
    for (int i = tid; i < BATCH * HSTRIDE; i += 512) sH[i] = (bf16_t)0.0f;
    __syncthreads();

    // per-lane gate biases (constant over t; added in the epilogue, NOT
    // splatted into accumulators -> no 8-wide bias vectors to spill)
    float biasv[4][4];
#pragma unroll
    for (int ci = 0; ci < 4; ++ci) {
        const int c = c0 + ci;
#pragma unroll
        for (int g = 0; g < 4; ++g) {
            const int col = g * HID + c * 16 + (lane & 15);
            biasv[ci][g] = b1[col] + b2[col];
        }
    }

    v8f cst[4];                                  // cell state, 4 tiles
#pragma unroll
    for (int ci = 0; ci < 4; ++ci)
#pragma unroll
        for (int p = 0; p < 8; ++p) cst[ci][p] = 0.0f;

    const bf16_t* inRing  = rings + (size_t)(l - 1) * ringSteps * BATCH * HID;
    bf16_t*       outRing = rings + (size_t)l * ringSteps * BATCH * HID;
    const bf16_t* hsrc    = sH + (r * 16) * HSTRIDE;

    for (int t = 0; t < T_SEQ; ++t) {
        // ---- phase 0: flow control (thread 0) ----
        if (tid == 0) {
            if (l > 0) wait_ge(&produced[l - 1], t + 1);               // input ready
            if (!lastLayer && t >= ringSteps)
                wait_ge(&consumed[l + 1], t - ringSteps + 1);          // slot free
        }
        __syncthreads();  // flags ok; also orders previous step's sH writes

        // ---- phase 1: load A fragments (x_t, h_{t-1}) ----
        v16bf xf[4], hf[4];
        if (firstLayer) {
            const float* xsrc = embIn + (size_t)t * (BATCH * HID) + r * 16 * HID;
#pragma unroll
            for (int kc = 0; kc < 4; ++kc) xf[kc] = load_a_f32cvt(xsrc, lane, kc * 32);
        } else {
            const bf16_t* xsrc = inRing + (size_t)(t & ringMask) * (BATCH * HID)
                               + r * 16 * HID;
#pragma unroll
            for (int kc = 0; kc < 4; ++kc) xf[kc] = load_a_bf16(xsrc, HID, lane, kc * 32);
        }
#pragma unroll
        for (int kc = 0; kc < 4; ++kc) hf[kc] = load_a_bf16(hsrc, HSTRIDE, lane, kc * 32);

        __syncthreads();  // all sH reads done before overwrite below
        if (tid == 0 && l > 0) signal(&consumed[l], t + 1);

        float*  outF = outFinal + (size_t)t * (BATCH * HID);
        bf16_t* outB = outRing + (size_t)(t & ringMask) * (BATCH * HID);

        // ---- phase 2: gates via WMMA + elementwise LSTM update ----
#pragma unroll
        for (int ci = 0; ci < 4; ++ci) {
            const int c = c0 + ci;
            v8f acc[4];
#pragma unroll
            for (int g = 0; g < 4; ++g)
#pragma unroll
                for (int p = 0; p < 8; ++p) acc[g][p] = 0.0f;  // inline-0 C

#pragma unroll
            for (int g = 0; g < 4; ++g) {
                const int ncol = g * HID + c * 16;
#pragma unroll
                for (int kc = 0; kc < 4; ++kc) {
                    v16bf bW = load_b_lds(sWih, ncol, lane, kc * 32);
                    acc[g] = __builtin_amdgcn_wmma_f32_16x16x32_bf16(
                        false, xf[kc], false, bW, (short)0, acc[g], false, false);
                    v16bf bU = load_b_lds(sWhh, ncol, lane, kc * 32);
                    acc[g] = __builtin_amdgcn_wmma_f32_16x16x32_bf16(
                        false, hf[kc], false, bU, (short)0, acc[g], false, false);
                }
            }

            const int hi8  = (lane & 16) ? 8 : 0;
            const int coln = c * 16 + (lane & 15);
            const float bi = biasv[ci][0], bf = biasv[ci][1];
            const float bg = biasv[ci][2], bo = biasv[ci][3];
#pragma unroll
            for (int p = 0; p < 8; ++p) {
                const float iv = fast_sigmoid(acc[0][p] + bi);
                const float fv = fast_sigmoid(acc[1][p] + bf);
                const float gv = fast_tanh(acc[2][p] + bg);
                const float ov = fast_sigmoid(acc[3][p] + bo);
                const float cn = fv * cst[ci][p] + iv * gv;
                cst[ci][p] = cn;
                const float hv = ov * fast_tanh(cn);
                const int row = r * 16 + hi8 + p;               // batch index
                if (lastLayer) outF[(size_t)row * HID + coln] = hv;      // f32 output
                else           outB[(size_t)row * HID + coln] = (bf16_t)hv; // bf16 ring
                sH[row * HSTRIDE + coln] = (bf16_t)hv;          // bf16 recurrence
            }
        }

        // ---- phase 3: make h_t visible device-wide, then release ----
        __threadfence();
        __syncthreads();
        if (tid == 0) signal(&produced[l], t + 1);
    }
}

// =====================================================================
extern "C" void kernel_launch(void* const* d_in, const int* in_sizes, int n_in,
                              void* d_out, int out_size, void* d_ws, size_t ws_size,
                              hipStream_t stream) {
    (void)in_sizes; (void)n_in; (void)out_size;

    const int*   x    = (const int*)d_in[0];
    const float* emb  = (const float*)d_in[1];
    const float* Wih  = (const float*)d_in[2];
    const float* Whh  = (const float*)d_in[3];
    const float* bih  = (const float*)d_in[4];
    const float* bhh  = (const float*)d_in[5];

    float* out = (float*)d_out;
    const size_t TBH = (size_t)T_SEQ * BATCH * HID;
    float* outEmb = out + TBH;   // second tuple element; also layer-0 input

    // workspace: [0,4096) flags, rest = 19 bf16 ring buffers (L2-resident)
    int*    flags = (int*)d_ws;
    bf16_t* rings = (bf16_t*)((char*)d_ws + 4096);
    const size_t perStep = (size_t)BATCH * HID * 2;                    // 32 KB bf16
    const size_t avail   = ws_size > 4096 ? ws_size - 4096 : 0;
    long steps = (long)(avail / ((size_t)(NLAYER - 1) * perStep));
    int ringSteps = 1;                                                 // pow2, <=128
    while ((long)ringSteps * 2 <= steps && ringSteps < 128) ringSteps <<= 1;

    zero_flags_kernel<<<1, 64, 0, stream>>>(flags);
    embed_kernel<<<(int)(TBH / 4 / 256), 256, 0, stream>>>(x, emb, outEmb);

    // LDS: 2 weight matrices [512][136] + h state [128][136], bf16
    const size_t smem = (size_t)(2 * G4H * WSTRIDE + BATCH * HSTRIDE) * 2;  // 313,344 B
    lstm_pipeline<<<NLAYER, 512, smem, stream>>>(outEmb, Wih, Whh, bih, bhh,
                                                 out, rings, ringSteps - 1, flags);
}
